// SpeechInput_19808389169903
// MI455X (gfx1250) — matrix operand order repaired and checked by reading, not compiled, not run on hardware
//
#include <hip/hip_runtime.h>

// ---------------------------------------------------------------------------
// Fused  out[t,b,o] = (src[b,:,t] @ W1^T + b1) @ W2^T + b2 + PE[t,o]
// for MI455X (gfx1250): bf16 WMMA 16x16x32, f32 accumulate, h-tile kept in LDS.
// GEMM2 software-pipelines the global W2 fragment load one k-step ahead to
// cover L2 latency at the low occupancy (1 WG/WGP) forced by the 290KB LDS tile.
// ---------------------------------------------------------------------------

typedef __attribute__((ext_vector_type(16))) __bf16 v16bf;
typedef __attribute__((ext_vector_type(8)))  __bf16 v8bf;
typedef __attribute__((ext_vector_type(8)))  float  v8f;

#define T_LEN    3000
#define N_BATCH  64
#define IN_DIM   80
#define H_DIM    1024
#define OUT_DIM  256
#define M_TILE   128
#define K1_PAD   96        // IN_DIM padded to 3 * 32
#define SRCA_PITCH 104     // halves; 208 B -> 13 16B-granules (odd => conflict-free)
#define H_PITCH    1032    // halves; 2064 B -> 129 granules (odd => conflict-free)

__device__ __forceinline__ v16bf cvt16(float4 f0, float4 f1, float4 f2, float4 f3) {
    union { v16bf v; __bf16 h[16]; } u;
    u.h[ 0] = (__bf16)f0.x; u.h[ 1] = (__bf16)f0.y; u.h[ 2] = (__bf16)f0.z; u.h[ 3] = (__bf16)f0.w;
    u.h[ 4] = (__bf16)f1.x; u.h[ 5] = (__bf16)f1.y; u.h[ 6] = (__bf16)f1.z; u.h[ 7] = (__bf16)f1.w;
    u.h[ 8] = (__bf16)f2.x; u.h[ 9] = (__bf16)f2.y; u.h[10] = (__bf16)f2.z; u.h[11] = (__bf16)f2.w;
    u.h[12] = (__bf16)f3.x; u.h[13] = (__bf16)f3.y; u.h[14] = (__bf16)f3.z; u.h[15] = (__bf16)f3.w;
    return u.v;
}

// Load 16 contiguous fp32 and convert to a bf16 B-fragment (contiguous K order used here, same as A:
// lane = N, halves j = K run of 16 starting at 32*kk + 16*(lane>>4)).
__device__ __forceinline__ v16bf load_cvt16(const float* __restrict__ p) {
    const float4* p4 = (const float4*)p;   // 64B-aligned by construction
    return cvt16(p4[0], p4[1], p4[2], p4[3]);
}

__device__ __forceinline__ v16bf zero16bf() {
    union { v16bf v; __bf16 h[16]; } u;
#pragma unroll
    for (int j = 0; j < 16; ++j) u.h[j] = (__bf16)0.0f;
    return u.v;
}

// A-fragment from an LDS row: two 16-byte runs, K = base+[0,8) and base+8+[0,8)
__device__ __forceinline__ v16bf load_a_lds(const __bf16* __restrict__ rowk) {
    union { v16bf v; v8bf r[2]; } u;
    u.r[0] = *(const v8bf*)(rowk);
    u.r[1] = *(const v8bf*)(rowk + 8);
    return u.v;
}

extern "C" __global__ __launch_bounds__(256)
void fused_mlp_pe_kernel(const float* __restrict__ src,
                         const float* __restrict__ W1,
                         const float* __restrict__ b1,
                         const float* __restrict__ W2,
                         const float* __restrict__ b2,
                         float* __restrict__ out) {
    extern __shared__ char smem_raw[];
    __bf16* sh_h   = (__bf16*)smem_raw;                                  // [128][H_PITCH]
    __bf16* sh_src = (__bf16*)(smem_raw + (size_t)M_TILE * H_PITCH * 2); // [128][SRCA_PITCH]

    const int tid  = threadIdx.x;
    const int wave = tid >> 5;        // 0..7
    const int lane = tid & 31;
    const int ln   = lane & 15;       // N column / M row selector within tile
    const int g    = lane >> 4;       // lane half-group
    const int m0   = blockIdx.x * M_TILE;

    // ---- Stage src tile into LDS as bf16, K padded 80 -> 96 with zeros.
    // idx = k*128 + r : consecutive tid -> consecutive r -> consecutive t
    // (src[b,k,t] is contiguous in t) => coalesced global reads.
    for (int idx = tid; idx < M_TILE * K1_PAD; idx += 256) {
        int k = idx >> 7;             // / 128
        int r = idx & 127;
        int m = m0 + r;
        int b = m / T_LEN;
        int t = m - b * T_LEN;
        float v = (k < IN_DIM) ? src[(b * IN_DIM + k) * T_LEN + t] : 0.0f;
        sh_src[r * SRCA_PITCH + k] = (__bf16)v;
    }
    __syncthreads();

    // ---- GEMM1: h[128 x 1024] = srcA @ W1^T (+ b1); wave w owns n-tiles [8w, 8w+8)
    for (int nti = 0; nti < 8; ++nti) {
        const int nt = wave * 8 + nti;
        const int n  = nt * 16 + ln;

        // Preload all 3 K-step B fragments up front (W1 row is K-contiguous).
        v16bf bfrag[3];
#pragma unroll
        for (int kk = 0; kk < 3; ++kk) {
            const int kb = kk * 32 + 16 * g;
            bfrag[kk] = (kb < IN_DIM) ? load_cvt16(&W1[n * IN_DIM + kb])
                                      : zero16bf();      // padded K region
        }

        v8f acc[8];
#pragma unroll
        for (int mt = 0; mt < 8; ++mt)
            acc[mt] = (v8f){0.f, 0.f, 0.f, 0.f, 0.f, 0.f, 0.f, 0.f};

#pragma unroll
        for (int kk = 0; kk < 3; ++kk) {
            const int ka = kk * 32 + 16 * g;
#pragma unroll
            for (int mt = 0; mt < 8; ++mt) {
                const __bf16* rowk = sh_src + (mt * 16 + ln) * SRCA_PITCH + ka;
                v16bf afrag = load_a_lds(rowk);
                acc[mt] = __builtin_amdgcn_wmma_f32_16x16x32_bf16(
                    false, afrag, false, bfrag[kk], (short)0, acc[mt], false, false);
            }
        }

        const float bias = b1[n];
#pragma unroll
        for (int mt = 0; mt < 8; ++mt) {
#pragma unroll
            for (int r = 0; r < 8; ++r) {
                int row = mt * 16 + r + 8 * g;             // C layout: M = r + 8g
                sh_h[row * H_PITCH + nt * 16 + ln] = (__bf16)(acc[mt][r] + bias);
            }
        }
    }
    __syncthreads();

    // ---- GEMM2: out_tile[128 x 256] = h @ W2^T (+ b2 + PE); wave owns 2 n-tiles.
    const float nlog = -0.036005772f;  // -ln(10000)/256
    for (int nti = 0; nti < 2; ++nti) {
        const int nt = wave * 2 + nti;
        const int o  = nt * 16 + ln;

        v8f acc[8];
#pragma unroll
        for (int mt = 0; mt < 8; ++mt)
            acc[mt] = (v8f){0.f, 0.f, 0.f, 0.f, 0.f, 0.f, 0.f, 0.f};

        // Software pipeline: prefetch kk+1's raw W2 run while WMMAs consume kk.
        const float4* w2p = (const float4*)&W2[o * H_DIM + 16 * g]; // +8 float4 per kk
        float4 nx0 = w2p[0], nx1 = w2p[1], nx2 = w2p[2], nx3 = w2p[3];

        for (int kk = 0; kk < 32; ++kk) {
            float4 c0 = nx0, c1 = nx1, c2 = nx2, c3 = nx3;
            if (kk < 31) {
                const float4* p = w2p + (kk + 1) * 8;
                nx0 = p[0]; nx1 = p[1]; nx2 = p[2]; nx3 = p[3];
            }
            v16bf bfrag = cvt16(c0, c1, c2, c3);
            const int ka = kk * 32 + 16 * g;
#pragma unroll
            for (int mt = 0; mt < 8; ++mt) {
                const __bf16* rowk = sh_h + (mt * 16 + ln) * H_PITCH + ka;
                v16bf afrag = load_a_lds(rowk);
                acc[mt] = __builtin_amdgcn_wmma_f32_16x16x32_bf16(
                    false, afrag, false, bfrag, (short)0, acc[mt], false, false);
            }
        }

        const float bias  = b2[o];
        const float rate  = __expf(nlog * (float)(o & ~1));
        const bool  isCos = (o & 1) != 0;
#pragma unroll
        for (int mt = 0; mt < 8; ++mt) {
#pragma unroll
            for (int r = 0; r < 8; ++r) {
                int m = m0 + mt * 16 + r + 8 * g;
                int b = m / T_LEN;
                int t = m - b * T_LEN;
                float ang = rate * (float)t;
                float pe  = isCos ? __cosf(ang) : __sinf(ang);
                out[(t * N_BATCH + b) * OUT_DIM + o] = acc[mt][r] + bias + pe;
            }
        }
    }
}

extern "C" void kernel_launch(void* const* d_in, const int* in_sizes, int n_in,
                              void* d_out, int out_size, void* d_ws, size_t ws_size,
                              hipStream_t stream) {
    (void)in_sizes; (void)n_in; (void)out_size; (void)d_ws; (void)ws_size;
    const float* src = (const float*)d_in[0];
    const float* W1  = (const float*)d_in[1];
    const float* b1  = (const float*)d_in[2];
    const float* W2  = (const float*)d_in[3];
    const float* b2  = (const float*)d_in[4];
    float* out = (float*)d_out;

    const size_t shmem = (size_t)(M_TILE * H_PITCH + M_TILE * SRCA_PITCH) * 2; // 290816 B
    const int grid = (N_BATCH * T_LEN) / M_TILE;  // 1500
    hipLaunchKernelGGL(fused_mlp_pe_kernel, dim3(grid), dim3(256), shmem, stream,
                       src, W1, b1, W2, b2, out);
}
